// LlamaMoeLayer_27582279975235
// MI455X (gfx1250) — matrix operand order repaired and checked by reading, not compiled, not run on hardware
//
#include <hip/hip_runtime.h>
#include <math.h>

// Problem constants (match reference)
#define H_    1024
#define I_    2048
#define T_    4096
#define E_    8
#define TWOI_ 4096

// Tiling: 256 threads = 8 waves in 4(M) x 2(N); each wave owns a 16x32 C strip
// (two 16x16 WMMA tiles) -> block tile 64(M) x 64(N), K-chunk 32 (bf16 WMMA K).
#define BM 64
#define BN 64
#define BK 32
// LDS row stride in halves: 36 halves = 72 bytes -> 8B-aligned packed stores,
// and odd multiple of 8 bytes per row to skew bank access.
#define LDA 36
#define LDB 36

typedef __attribute__((ext_vector_type(16))) __bf16 v16bf;
typedef __attribute__((ext_vector_type(8)))  float  v8f;

union BF16x16 { v16bf v; __bf16 e[16]; };
union F32x8   { v8f   v; float  e[8];  };

// Native conversion: clang emits packed v_cvt_pk_bf16_f32 (RNE) on gfx1250.
__device__ __forceinline__ __bf16 f2bf(float f) { return (__bf16)f; }

// pack 4 floats -> 4 bf16, single 8-byte store (dst must be 8B aligned)
__device__ __forceinline__ void st4bf(__bf16* dst, float4 v) {
  union { __bf16 b[4]; unsigned long long u; } p;
  p.b[0] = f2bf(v.x); p.b[1] = f2bf(v.y); p.b[2] = f2bf(v.z); p.b[3] = f2bf(v.w);
  *(unsigned long long*)dst = p.u;
}

// Fast sigmoid via hardware v_rcp_f32 (~1 ulp) instead of IEEE div expansion.
__device__ __forceinline__ float fast_sigmoid(float x) {
  return __builtin_amdgcn_rcpf(1.f + __expf(-x));
}

// ---- WMMA fragment loads per ISA 16-bit VGPR layouts (wave32) --------------
__device__ __forceinline__ v16bf load_a_frag(const __bf16* As, int wm, int lc, int lh) {
  BF16x16 a;
  const __bf16* ar = &As[(wm * 16 + lc) * LDA];
  #pragma unroll
  for (int i = 0; i < 8; ++i) {
    a.e[i]     = ar[lh * 8 + i];        // K = 0..7   (+8 for hi half-wave)
    a.e[8 + i] = ar[16 + lh * 8 + i];   // K = 16..23 (+8 for hi half-wave)
  }
  return a.v;
}
__device__ __forceinline__ v16bf load_b_frag(const __bf16* Bs, int ncol, int lh) {
  BF16x16 b;
  const __bf16* br = &Bs[ncol * LDB];
  #pragma unroll
  for (int i = 0; i < 16; ++i) b.e[i] = br[(i >> 3) * 16 + lh * 8 + (i & 7)];  // K = 0..7,16..23 (+8 hi)
  return b.v;
}

// ---------------------------------------------------------------- router ----
__global__ __launch_bounds__(128)
void init_counts_kernel(int* counts) {
  if (threadIdx.x < E_) counts[threadIdx.x] = 0;
}

__global__ __launch_bounds__(128)
void router_kernel(const float* __restrict__ X, const float* __restrict__ RW,
                   float* __restrict__ logits_out,
                   int* __restrict__ counts, int* __restrict__ tok,
                   float* __restrict__ sc) {
  const int t = blockIdx.x;
  __shared__ float red[E_][129];
  __shared__ float lsum[E_];
  float acc[E_];
  #pragma unroll
  for (int e = 0; e < E_; ++e) acc[e] = 0.f;
  for (int h = threadIdx.x; h < H_; h += 128) {
    float xv = X[(size_t)t * H_ + h];
    #pragma unroll
    for (int e = 0; e < E_; ++e) acc[e] += xv * RW[e * H_ + h];
  }
  #pragma unroll
  for (int e = 0; e < E_; ++e) red[e][threadIdx.x] = acc[e];
  __syncthreads();
  if (threadIdx.x < E_) {
    const int e = threadIdx.x;
    float s = 0.f;
    for (int i = 0; i < 128; ++i) s += red[e][i];
    lsum[e] = s;
    logits_out[(size_t)t * E_ + e] = s;
  }
  __syncthreads();
  if (threadIdx.x == 0) {
    float l[E_];
    #pragma unroll
    for (int e = 0; e < E_; ++e) l[e] = lsum[e];
    int i1 = 0;
    #pragma unroll
    for (int e = 1; e < E_; ++e) if (l[e] > l[i1]) i1 = e;
    int i2 = (i1 == 0) ? 1 : 0;
    #pragma unroll
    for (int e = 0; e < E_; ++e) if (e != i1 && l[e] > l[i2]) i2 = e;
    float s1 = fast_sigmoid(l[i1]);
    float s2 = fast_sigmoid(l[i2]);
    int p1 = atomicAdd(&counts[i1], 1);
    tok[i1 * T_ + p1] = t;  sc[i1 * T_ + p1] = s1;
    int p2 = atomicAdd(&counts[i2], 1);
    tok[i2 * T_ + p2] = t;  sc[i2 * T_ + p2] = s2;
  }
}

__global__ __launch_bounds__(32)
void prefix_kernel(const int* __restrict__ counts, int* __restrict__ offs) {
  if (threadIdx.x == 0) {
    int s = 0;
    for (int e = 0; e < E_; ++e) { offs[e] = s; s += counts[e]; }
    offs[E_] = s;   // == 2*T
  }
}

// --------------------------------------------------- fused gate+up GEMM ----
// MODE 0: shared expert  (A rows = tokens, scale 1, B = shared_gate/up [I,H])
// MODE 1: routed experts (A rows gathered & score-scaled, B = gate_up[e][H,2I])
// Epilogue writes h = up * silu(gate) as bf16.
template <int MODE>
__global__ __launch_bounds__(256)
void moe_gateup_kernel(const float* __restrict__ X,
                       const float* __restrict__ GUP,
                       const float* __restrict__ SGW,
                       const float* __restrict__ SUW,
                       const int* __restrict__ counts,
                       const int* __restrict__ offs,
                       const int* __restrict__ tok,
                       const float* __restrict__ sc,
                       void* __restrict__ hout_) {
  __shared__ __bf16 As[BM * LDA];
  __shared__ __bf16 Bg[BN * LDB];   // transposed: [n][k]
  __shared__ __bf16 Bu[BN * LDB];

  const int n0 = blockIdx.x * BN;   // column within I (gate; up pairs at +I)
  const int m0 = blockIdx.y * BM;

  int cnt = T_, rowbase = m0;
  const int*   tokE = nullptr;
  const float* scE  = nullptr;
  const float* gupE = nullptr;
  if (MODE) {
    const int e = blockIdx.z;
    cnt = counts[e];
    if (m0 >= cnt) return;                 // block-uniform early exit
    rowbase = offs[e] + m0;
    tokE = tok + e * T_;
    scE  = sc  + e * T_;
    gupE = GUP + (size_t)e * H_ * TWOI_;
  }
  __bf16* hout = (__bf16*)hout_;

  const int tid  = threadIdx.x;
  const int lane = tid & 31;               // wave32
  const int wid  = tid >> 5;
  const int wm   = wid & 3;                // M sub-tile
  const int wn   = wid >> 2;               // N sub-strip (32 wide)
  const int lh   = lane >> 4;
  const int lc   = lane & 15;

  // ---- A staging map: 2 rows x float4 per thread. Branchless gather: clamp
  // out-of-range rows to a valid list entry and zero their scale (score
  // pre-scaling makes the padded rows contribute exactly zero).
  const int ar0 = tid >> 3;                // 0..31 (second row: +32)
  const int ak  = (tid & 7) * 4;           // 0,4,...,28
  const float* arow[2];
  float ascale[2] = {1.f, 1.f};
  #pragma unroll
  for (int p = 0; p < 2; ++p) {
    const int r = ar0 + p * 32;
    if (MODE) {
      const int rr   = m0 + r;
      const bool ok  = rr < cnt;
      const int idx  = ok ? rr : (cnt - 1);        // cnt >= 1 here
      arow[p]   = X + (size_t)tokE[idx] * H_;
      ascale[p] = ok ? scE[idx] : 0.f;
    } else {
      arow[p] = X + (size_t)(m0 + r) * H_;
    }
  }

  // ---- B staging maps
  const int bk1 = tid >> 4;                // MODE1: k row 0..15 (+16)
  const int bn1 = (tid & 15) * 4;          // MODE1: 4 consecutive n
  const int bn0 = tid >> 2;                // MODE0: n row 0..63
  const int bk0 = (tid & 3) * 8;           // MODE0: 8 consecutive k

  F32x8 acc[4];                            // [gate0, gate1, up0, up1]
  #pragma unroll
  for (int q = 0; q < 4; ++q)
    #pragma unroll
    for (int i = 0; i < 8; ++i) acc[q].e[i] = 0.f;

  for (int k0 = 0; k0 < H_; k0 += BK) {
    __syncthreads();
    // ---- stage A (f32 -> bf16, pre-scaled, branchless)
    #pragma unroll
    for (int p = 0; p < 2; ++p) {
      const int r = ar0 + p * 32;
      float4 xv = *(const float4*)(arow[p] + k0 + ak);
      if (MODE) {
        xv.x *= ascale[p]; xv.y *= ascale[p];
        xv.z *= ascale[p]; xv.w *= ascale[p];
      }
      st4bf(&As[r * LDA + ak], xv);
    }
    // ---- stage B (f32 -> bf16, transposed into LDS)
    if (MODE) {                            // [H,2I]: contiguous in n
      #pragma unroll
      for (int p = 0; p < 2; ++p) {
        const int k = bk1 + p * 16;
        const float* pg = gupE + (size_t)(k0 + k) * TWOI_ + (n0 + bn1);
        float4 g4 = *(const float4*)pg;
        float4 u4 = *(const float4*)(pg + I_);
        Bg[(bn1 + 0) * LDB + k] = f2bf(g4.x);
        Bg[(bn1 + 1) * LDB + k] = f2bf(g4.y);
        Bg[(bn1 + 2) * LDB + k] = f2bf(g4.z);
        Bg[(bn1 + 3) * LDB + k] = f2bf(g4.w);
        Bu[(bn1 + 0) * LDB + k] = f2bf(u4.x);
        Bu[(bn1 + 1) * LDB + k] = f2bf(u4.y);
        Bu[(bn1 + 2) * LDB + k] = f2bf(u4.z);
        Bu[(bn1 + 3) * LDB + k] = f2bf(u4.w);
      }
      if (k0 + BK < H_)
        __builtin_prefetch(gupE + (size_t)(k0 + BK) * TWOI_ + n0, 0, 1);
    } else {                               // [I,H]: contiguous in k
      const float* pg = SGW + (size_t)(n0 + bn0) * H_ + k0 + bk0;
      const float* pu = SUW + (size_t)(n0 + bn0) * H_ + k0 + bk0;
      st4bf(&Bg[bn0 * LDB + bk0],     *(const float4*)pg);
      st4bf(&Bg[bn0 * LDB + bk0 + 4], *(const float4*)(pg + 4));
      st4bf(&Bu[bn0 * LDB + bk0],     *(const float4*)pu);
      st4bf(&Bu[bn0 * LDB + bk0 + 4], *(const float4*)(pu + 4));
    }
    __syncthreads();

    // ---- compute: 4 WMMA per wave per K-chunk
    v16bf a   = load_a_frag(As, wm, lc, lh);
    v16bf bg0 = load_b_frag(Bg, wn * 32 + lc,      lh);
    v16bf bg1 = load_b_frag(Bg, wn * 32 + 16 + lc, lh);
    v16bf bu0 = load_b_frag(Bu, wn * 32 + lc,      lh);
    v16bf bu1 = load_b_frag(Bu, wn * 32 + 16 + lc, lh);
    acc[0].v = __builtin_amdgcn_wmma_f32_16x16x32_bf16(false, a, false, bg0, (short)0, acc[0].v, false, false);
    acc[1].v = __builtin_amdgcn_wmma_f32_16x16x32_bf16(false, a, false, bg1, (short)0, acc[1].v, false, false);
    acc[2].v = __builtin_amdgcn_wmma_f32_16x16x32_bf16(false, a, false, bu0, (short)0, acc[2].v, false, false);
    acc[3].v = __builtin_amdgcn_wmma_f32_16x16x32_bf16(false, a, false, bu1, (short)0, acc[3].v, false, false);
  }

  // ---- epilogue: h = up * silu(gate), two 16-wide column groups
  const int nA = n0 + wn * 32 + lc;
  const int nB = nA + 16;
  #pragma unroll
  for (int i = 0; i < 8; ++i) {
    const int rl = wm * 16 + lh * 8 + i;
    if (MODE && (m0 + rl >= cnt)) continue;
    float g0 = acc[0].e[i], u0 = acc[2].e[i];
    float g1 = acc[1].e[i], u1 = acc[3].e[i];
    float h0 = u0 * g0 * fast_sigmoid(g0);
    float h1 = u1 * g1 * fast_sigmoid(g1);
    __bf16* hr = hout + (size_t)(rowbase + rl) * I_;
    hr[nA] = f2bf(h0);
    hr[nB] = f2bf(h1);
  }
}

// -------------------------------------------------------------- down GEMM ---
// MODE 0: shared expert, plain store (fully initializes d_out).
// MODE 1: routed experts, scatter-accumulate via global f32 atomics.
template <int MODE>
__global__ __launch_bounds__(256)
void moe_down_kernel(const void* __restrict__ Hin_,
                     const float* __restrict__ DWN,
                     const float* __restrict__ SDW,
                     const int* __restrict__ counts,
                     const int* __restrict__ offs,
                     const int* __restrict__ tok,
                     float* __restrict__ OUT) {
  __shared__ __bf16 As[BM * LDA];
  __shared__ __bf16 Bs[BN * LDB];

  const int n0 = blockIdx.x * BN;          // column within H
  const int m0 = blockIdx.y * BM;

  int cnt = T_, rowbase = m0;
  const int*   tokE = nullptr;
  const float* dwnE = nullptr;
  if (MODE) {
    const int e = blockIdx.z;
    cnt = counts[e];
    if (m0 >= cnt) return;
    rowbase = offs[e] + m0;
    tokE = tok + e * T_;
    dwnE = DWN + (size_t)e * I_ * H_;
  }
  const __bf16* Hin = (const __bf16*)Hin_;

  const int tid  = threadIdx.x;
  const int lane = tid & 31;
  const int wid  = tid >> 5;
  const int wm   = wid & 3;
  const int wn   = wid >> 2;
  const int lh   = lane >> 4;
  const int lc   = lane & 15;

  // A staging: 2 rows x 4 bf16 (8 bytes) per thread; rows are contiguous in
  // the compacted activation buffer. Branchless: clamp OOR rows to the first
  // row of this (valid) tile and mask the value to zero.
  const int ar0 = tid >> 3;
  const int ak  = (tid & 7) * 4;
  int  arowi[2];
  bool aval[2];
  #pragma unroll
  for (int p = 0; p < 2; ++p) {
    const int r = ar0 + p * 32;
    aval[p]  = (!MODE) || (m0 + r < cnt);
    arowi[p] = rowbase + (aval[p] ? r : 0);
  }
  const int bk1 = tid >> 4;
  const int bn1 = (tid & 15) * 4;
  const int bn0 = tid >> 2;
  const int bk0 = (tid & 3) * 8;

  F32x8 acc0, acc1;
  #pragma unroll
  for (int i = 0; i < 8; ++i) { acc0.e[i] = 0.f; acc1.e[i] = 0.f; }

  for (int k0 = 0; k0 < I_; k0 += BK) {
    __syncthreads();
    #pragma unroll
    for (int p = 0; p < 2; ++p) {
      const int r = ar0 + p * 32;
      unsigned long long v =
          *(const unsigned long long*)(Hin + (size_t)arowi[p] * I_ + k0 + ak);
      if (MODE) v = aval[p] ? v : 0ull;
      *(unsigned long long*)&As[r * LDA + ak] = v;
    }
    if (MODE) {                            // [I,H]: contiguous in n
      #pragma unroll
      for (int p = 0; p < 2; ++p) {
        const int k = bk1 + p * 16;
        float4 w4 = *(const float4*)(dwnE + (size_t)(k0 + k) * H_ + (n0 + bn1));
        Bs[(bn1 + 0) * LDB + k] = f2bf(w4.x);
        Bs[(bn1 + 1) * LDB + k] = f2bf(w4.y);
        Bs[(bn1 + 2) * LDB + k] = f2bf(w4.z);
        Bs[(bn1 + 3) * LDB + k] = f2bf(w4.w);
      }
      if (k0 + BK < I_)
        __builtin_prefetch(dwnE + (size_t)(k0 + BK) * H_ + n0, 0, 1);
    } else {                               // [H,I]: contiguous in k
      const float* pw = SDW + (size_t)(n0 + bn0) * I_ + k0 + bk0;
      st4bf(&Bs[bn0 * LDB + bk0],     *(const float4*)pw);
      st4bf(&Bs[bn0 * LDB + bk0 + 4], *(const float4*)(pw + 4));
    }
    __syncthreads();

    v16bf a  = load_a_frag(As, wm, lc, lh);
    v16bf b0 = load_b_frag(Bs, wn * 32 + lc,      lh);
    v16bf b1 = load_b_frag(Bs, wn * 32 + 16 + lc, lh);
    acc0.v = __builtin_amdgcn_wmma_f32_16x16x32_bf16(false, a, false, b0, (short)0, acc0.v, false, false);
    acc1.v = __builtin_amdgcn_wmma_f32_16x16x32_bf16(false, a, false, b1, (short)0, acc1.v, false, false);
  }

  const int nA = n0 + wn * 32 + lc;
  const int nB = nA + 16;
  #pragma unroll
  for (int i = 0; i < 8; ++i) {
    const int rl = wm * 16 + lh * 8 + i;
    if (MODE) {
      if (m0 + rl < cnt) {
        float* orow = OUT + (size_t)tokE[m0 + rl] * H_;
        atomicAdd(orow + nA, acc0.e[i]);
        atomicAdd(orow + nB, acc1.e[i]);
      }
    } else {
      float* orow = OUT + (size_t)(m0 + rl) * H_;
      orow[nA] = acc0.e[i];
      orow[nB] = acc1.e[i];
    }
  }
}

// ------------------------------------------------------------------ host ----
extern "C" void kernel_launch(void* const* d_in, const int* in_sizes, int n_in,
                              void* d_out, int out_size, void* d_ws, size_t ws_size,
                              hipStream_t stream) {
  const float* X   = (const float*)d_in[0];   // [T,H]
  const float* RW  = (const float*)d_in[1];   // [E,H]
  const float* GUP = (const float*)d_in[2];   // [E,H,2I]
  const float* DWN = (const float*)d_in[3];   // [E,I,H]
  const float* SGW = (const float*)d_in[4];   // [I,H]
  const float* SUW = (const float*)d_in[5];   // [I,H]
  const float* SDW = (const float*)d_in[6];   // [H,I]

  float* OUT    = (float*)d_out;              // [T,H] then logits [T,E]
  float* LOGITS = OUT + (size_t)T_ * H_;

  char* ws = (char*)d_ws;
  int*   counts = (int*)(ws);                                   // 8 ints
  int*   offs   = (int*)(ws + 64);                              // 9 ints
  int*   tok    = (int*)(ws + 1024);                            // E*T ints
  float* sc     = (float*)(ws + 1024 + sizeof(int) * E_ * T_);  // E*T floats
  char*  hsh    = ws + 1024 + 8 * E_ * T_;                      // bf16 [T,I]
  char*  hro    = hsh + 2ull * T_ * I_;                         // bf16 [2T,I]

  init_counts_kernel<<<dim3(1), dim3(128), 0, stream>>>(counts);
  router_kernel<<<dim3(T_), dim3(128), 0, stream>>>(X, RW, LOGITS, counts, tok, sc);
  prefix_kernel<<<dim3(1), dim3(32), 0, stream>>>(counts, offs);

  // Shared expert: gate+up fused, then down (plain stores initialize d_out).
  moe_gateup_kernel<0><<<dim3(I_ / BN, T_ / BM, 1), dim3(256), 0, stream>>>(
      X, GUP, SGW, SUW, counts, offs, tok, sc, (void*)hsh);
  moe_down_kernel<0><<<dim3(H_ / BN, T_ / BM, 1), dim3(256), 0, stream>>>(
      (const void*)hsh, DWN, SDW, counts, offs, tok, OUT);

  // Routed experts: sparse grouped GEMMs over gathered, score-scaled tokens.
  moe_gateup_kernel<1><<<dim3(I_ / BN, T_ / BM, E_), dim3(256), 0, stream>>>(
      X, GUP, SGW, SUW, counts, offs, tok, sc, (void*)hro);
  moe_down_kernel<1><<<dim3(H_ / BN, T_ / BM, E_), dim3(256), 0, stream>>>(
      (const void*)hro, DWN, SDW, counts, offs, tok, OUT);
}
